// APPNPNet_90580860272650
// MI455X (gfx1250) — matrix operand order repaired and checked by reading, not compiled, not run on hardware
//
#include <hip/hip_runtime.h>
#include <hip/hip_bf16.h>
#include <stdint.h>

#define ALPHA_F 0.1f
#define ONE_MINUS_ALPHA 0.9f
#define K_STEPS 10

typedef __bf16 bf16_t;
typedef __attribute__((ext_vector_type(16))) __bf16 v16bf;
typedef __attribute__((ext_vector_type(8)))  __bf16 v8bf;
typedef __attribute__((ext_vector_type(8)))  float  v8f;
typedef __attribute__((ext_vector_type(4)))  int    v4i;

#if defined(__has_builtin)
#if __has_builtin(__builtin_amdgcn_global_load_async_to_lds_b128)
#define HAVE_ASYNC_LDS 1
#endif
#endif
#ifndef HAVE_ASYNC_LDS
#define HAVE_ASYNC_LDS 0
#endif

#if HAVE_ASYNC_LDS
typedef __attribute__((address_space(1))) v4i* as1_v4i;
typedef __attribute__((address_space(3))) v4i* as3_v4i;

static __device__ __forceinline__ void async_cp16(const bf16_t* g, bf16_t* l) {
  // GLOBAL_LOAD_ASYNC_TO_LDS_B128: 16B per lane, tracked by ASYNCcnt
  __builtin_amdgcn_global_load_async_to_lds_b128((as1_v4i)(g), (as3_v4i)(l), 0, 0);
}
static __device__ __forceinline__ void async_wait0() {
#if __has_builtin(__builtin_amdgcn_s_wait_asynccnt)
  __builtin_amdgcn_s_wait_asynccnt(0);
#else
  asm volatile("s_wait_asynccnt 0x0" ::: "memory");
#endif
}
#endif

static __device__ __forceinline__ bf16_t f2bf(float f) {
  // round-to-nearest-even fp32 -> bf16
  unsigned u = __builtin_bit_cast(unsigned, f);
  u = (u + 0x7FFFu + ((u >> 16) & 1u)) >> 16;
  unsigned short s = (unsigned short)u;
  return __builtin_bit_cast(bf16_t, s);
}

static __device__ __forceinline__ v16bf cat16(v8bf lo, v8bf hi) {
  return __builtin_shufflevector(lo, hi, 0, 1, 2, 3, 4, 5, 6, 7,
                                         8, 9, 10, 11, 12, 13, 14, 15);
}

// ---------------------------------------------------------------------------
// One-shot fp32 -> bf16 conversion (bandwidth bound)
// ---------------------------------------------------------------------------
__global__ void f32_to_bf16(const float* __restrict__ in,
                            bf16_t* __restrict__ out, long long n) {
  long long i = (long long)blockIdx.x * blockDim.x + threadIdx.x;
  if (i < n) out[i] = f2bf(in[i]);
}

// ---------------------------------------------------------------------------
// Fused MLP: h = relu(x @ W1^T + b1) @ W2^T + b2      (bf16 WMMA, f32 accum)
// Block = 256 threads = 8 waves; block covers 128 rows.
// W1 double-buffered in LDS; staged via async global->LDS DMA (ASYNCcnt)
// when available, register relay otherwise. DMA/loads overlap the WMMAs.
// ---------------------------------------------------------------------------
__global__ __launch_bounds__(256) void mlp_fused(
    const bf16_t* __restrict__ xb,   // [N,512] bf16
    const bf16_t* __restrict__ W1b,  // [256,512] bf16 row-major
    const float*  __restrict__ b1,   // [256]
    const bf16_t* __restrict__ W2b,  // [16,256] bf16 row-major
    const float*  __restrict__ b2,   // [16]
    float* __restrict__ h,           // [N,16]
    int N)
{
  __shared__ bf16_t w1s[2][256][40]; // [buf][n][k] k-tile of 32 + 8 pad (80B rows)
  __shared__ bf16_t h1[128][264];    // 256 cols + 8 pad (16B-aligned rows)

  const int tid  = threadIdx.x;
  const int lane = tid & 31;
  const int wave = tid >> 5;
  const int rowBase = blockIdx.x * 128 + wave * 16;

  const int m    = lane & 15;        // row (A) / col (B) within 16
  const int aOff = (lane >> 4) << 3; // A-matrix K offset per half-wave (0 or 8)
  const int bOff = (lane >> 4) << 4; // unused; B loads also use aOff (0 or 8)

  const v8f vzero = {0.f, 0.f, 0.f, 0.f, 0.f, 0.f, 0.f, 0.f};
  v8f acc[16];
#pragma unroll
  for (int t = 0; t < 16; ++t) acc[t] = vzero;

  int arow = rowBase + m;
  if (arow >= N) arow = N - 1;       // clamp; stores guarded later
  const bf16_t* xrow = xb + (long long)arow * 512;
  const bf16_t* w1row = W1b + tid * 512;   // staging: thread tid owns W1 row tid

  // ---- prologue: stage k-tile 0, preload A fragment 0 --------------------
#if HAVE_ASYNC_LDS
#pragma unroll
  for (int c = 0; c < 4; ++c)
    async_cp16(w1row + c * 8, &w1s[0][tid][c * 8]);
#else
  {
    v8bf w0 = *(const v8bf*)(w1row);
    v8bf w1 = *(const v8bf*)(w1row + 8);
    v8bf w2 = *(const v8bf*)(w1row + 16);
    v8bf w3 = *(const v8bf*)(w1row + 24);
    *(v8bf*)&w1s[0][tid][0]  = w0;
    *(v8bf*)&w1s[0][tid][8]  = w1;
    *(v8bf*)&w1s[0][tid][16] = w2;
    *(v8bf*)&w1s[0][tid][24] = w3;
  }
#endif
  v8bf a0 = *(const v8bf*)(xrow + aOff);
  v8bf a1 = *(const v8bf*)(xrow + aOff + 16);
#if HAVE_ASYNC_LDS
  async_wait0();
#endif
  __syncthreads();

  // ---- GEMM1 main loop over 16 k-tiles -----------------------------------
#pragma unroll 1
  for (int kt = 0; kt < 16; ++kt) {
    const int k0  = kt * 32;
    const int buf = kt & 1;

    v8bf na0, na1;
#if !HAVE_ASYNC_LDS
    v8bf nw0, nw1, nw2, nw3;
#endif
    if (kt < 15) {                   // kick off next k-tile while we compute
      const bf16_t* wp = w1row + k0 + 32;
#if HAVE_ASYNC_LDS
#pragma unroll
      for (int c = 0; c < 4; ++c)
        async_cp16(wp + c * 8, &w1s[buf ^ 1][tid][c * 8]);
#else
      nw0 = *(const v8bf*)(wp);
      nw1 = *(const v8bf*)(wp + 8);
      nw2 = *(const v8bf*)(wp + 16);
      nw3 = *(const v8bf*)(wp + 24);
#endif
      na0 = *(const v8bf*)(xrow + k0 + 32 + aOff);
      na1 = *(const v8bf*)(xrow + k0 + 32 + aOff + 16);
    }

    v16bf a = cat16(a0, a1);
    // B fragments from LDS, pipelined one tile ahead
    v8bf bl = *(const v8bf*)&w1s[buf][m][aOff];
    v8bf bh = *(const v8bf*)&w1s[buf][m][aOff + 16];
#pragma unroll
    for (int t = 0; t < 16; ++t) {
      v8bf nbl, nbh;
      if (t < 15) {
        nbl = *(const v8bf*)&w1s[buf][(t + 1) * 16 + m][aOff];
        nbh = *(const v8bf*)&w1s[buf][(t + 1) * 16 + m][aOff + 16];
      }
      acc[t] = __builtin_amdgcn_wmma_f32_16x16x32_bf16(
          false, a, false, cat16(bl, bh), (short)0, acc[t], false, false);
      if (t < 15) { bl = nbl; bh = nbh; }
    }

    if (kt < 15) {
#if HAVE_ASYNC_LDS
      async_wait0();                 // DMA for buf^1 complete in this wave
#else
      const int nb = buf ^ 1;
      *(v8bf*)&w1s[nb][tid][0]  = nw0;
      *(v8bf*)&w1s[nb][tid][8]  = nw1;
      *(v8bf*)&w1s[nb][tid][16] = nw2;
      *(v8bf*)&w1s[nb][tid][24] = nw3;
#endif
      a0 = na0; a1 = na1;
    }
    __syncthreads();
  }

  // bias + ReLU, stash bf16 into LDS. C/D layout: VGPR r -> M = r + 8*(lane/16),
  // N = lane%16.
  const int cRow = (lane >> 4) << 3;
  const int cCol = lane & 15;
#pragma unroll
  for (int t = 0; t < 16; ++t) {
    float bias = b1[t * 16 + cCol];
#pragma unroll
    for (int r = 0; r < 8; ++r) {
      float v = acc[t][r] + bias;
      v = v > 0.f ? v : 0.f;
      h1[wave * 16 + cRow + r][t * 16 + cCol] = f2bf(v);
    }
  }
  __syncthreads();

  // ---- GEMM2: [16x256] x [256x16] ----------------------------------------
  v8f c2 = vzero;
#pragma unroll 1
  for (int k0 = 0; k0 < 256; k0 += 32) {
    v8bf g0 = *(const v8bf*)&h1[wave * 16 + m][k0 + aOff];        // ds_load_b128
    v8bf g1 = *(const v8bf*)&h1[wave * 16 + m][k0 + aOff + 16];
    const bf16_t* w2row = W2b + m * 256 + k0 + aOff;              // B2[k][n]=W2[n][k]
    v8bf bl = *(const v8bf*)(w2row);
    v8bf bh = *(const v8bf*)(w2row + 16);
    c2 = __builtin_amdgcn_wmma_f32_16x16x32_bf16(
        false, cat16(g0, g1), false, cat16(bl, bh), (short)0, c2, false, false);
  }

  float bias2 = b2[cCol];
#pragma unroll
  for (int r = 0; r < 8; ++r) {
    int row = rowBase + cRow + r;
    if (row < N) h[(long long)row * 16 + cCol] = c2[r] + bias2;
  }
}

// ---------------------------------------------------------------------------
// Graph prep
// ---------------------------------------------------------------------------
__global__ void deg_init(float* __restrict__ deg, int N) {
  int i = blockIdx.x * blockDim.x + threadIdx.x;
  if (i < N) deg[i] = 1.0f;                 // self-loop
}

__global__ void deg_accum(const long long* __restrict__ dst,
                          float* __restrict__ deg, int E) {
  int e = blockIdx.x * blockDim.x + threadIdx.x;
  if (e < E) atomicAdd(&deg[(int)dst[e]], 1.0f);
}

__global__ void dinv_finish(float* __restrict__ d, int N) {
  int i = blockIdx.x * blockDim.x + threadIdx.x;
  if (i < N) d[i] = rsqrtf(d[i]);           // deg >= 1 always (self-loop)
}

__global__ void edge_prep(const long long* __restrict__ ei,
                          int* __restrict__ src32, int* __restrict__ dst32,
                          float* __restrict__ norm,
                          const float* __restrict__ dinv, int E) {
  int e = blockIdx.x * blockDim.x + threadIdx.x;
  if (e < E) {
    int s = (int)ei[e];
    int d = (int)ei[(long long)E + e];
    src32[e] = s;
    dst32[e] = d;
    norm[e] = dinv[s] * dinv[d];
  }
}

// ---------------------------------------------------------------------------
// Propagation: z_next = (1-a) * A_hat z + a * h
// ---------------------------------------------------------------------------
__global__ void prop_init(const float* __restrict__ h,
                          const float* __restrict__ z,
                          const float* __restrict__ dinv,
                          float* __restrict__ out, int N) {
  long long t = (long long)blockIdx.x * blockDim.x + threadIdx.x;
  int i = (int)(t >> 4);
  if (i < N) {
    float di = dinv[i];
    out[t] = ALPHA_F * h[t] + ONE_MINUS_ALPHA * di * di * z[t];
  }
}

__global__ void prop_scatter(const int* __restrict__ src,
                             const int* __restrict__ dst,
                             const float* __restrict__ norm,
                             const float* __restrict__ z,
                             float* __restrict__ out, int E) {
  long long t = (long long)blockIdx.x * blockDim.x + threadIdx.x;
  int e = (int)(t >> 4);
  int c = (int)(t & 15);
  if (e < E) {
    float w = ONE_MINUS_ALPHA * norm[e];
    atomicAdd(&out[(long long)dst[e] * 16 + c], w * z[(long long)src[e] * 16 + c]);
  }
}

// ---------------------------------------------------------------------------
// log_softmax over 16 classes
// ---------------------------------------------------------------------------
__global__ void logsoftmax16(const float* __restrict__ z,
                             float* __restrict__ out, int N) {
  int i = blockIdx.x * blockDim.x + threadIdx.x;
  if (i < N) {
    const float* row = z + (long long)i * 16;
    float m = row[0];
#pragma unroll
    for (int c = 1; c < 16; ++c) m = fmaxf(m, row[c]);
    float s = 0.f;
#pragma unroll
    for (int c = 0; c < 16; ++c) s += __expf(row[c] - m);
    float l = __logf(s);
    float* orow = out + (long long)i * 16;
#pragma unroll
    for (int c = 0; c < 16; ++c) orow[c] = row[c] - m - l;
  }
}

// ---------------------------------------------------------------------------
extern "C" void kernel_launch(void* const* d_in, const int* in_sizes, int n_in,
                              void* d_out, int out_size, void* d_ws, size_t ws_size,
                              hipStream_t stream) {
  const float*     x  = (const float*)d_in[0];
  const float*     W1 = (const float*)d_in[1];
  const float*     b1 = (const float*)d_in[2];
  const float*     W2 = (const float*)d_in[3];
  const float*     b2 = (const float*)d_in[4];
  const long long* ei = (const long long*)d_in[5];

  const int N = in_sizes[0] / 512;
  const int E = in_sizes[5] / 2;
  float* out = (float*)d_out;

  // workspace layout
  float* h    = (float*)d_ws;               // N*16
  float* zA   = h    + (size_t)N * 16;      // N*16
  float* zB   = zA   + (size_t)N * 16;      // N*16
  float* dinv = zB   + (size_t)N * 16;      // N   (deg, then rsqrt(deg))
  float* norm = dinv + (size_t)N;           // E
  int* src32  = (int*)(norm + (size_t)E);   // E
  int* dst32  = src32 + (size_t)E;          // E
  uintptr_t p = (uintptr_t)(dst32 + (size_t)E);
  p = (p + 63) & ~(uintptr_t)63;            // 64B align for b128 vector loads
  bf16_t* W1b = (bf16_t*)p;                 // 256*512
  bf16_t* W2b = W1b + 256 * 512;            // 16*256
  bf16_t* xb  = W2b + 16 * 256;             // N*512 (still 64B-aligned)

  // Pre-convert operands to bf16
  const long long nx = (long long)N * 512;
  f32_to_bf16<<<(unsigned)((nx + 255) / 256), 256, 0, stream>>>(x, xb, nx);
  f32_to_bf16<<<(256 * 512 + 255) / 256, 256, 0, stream>>>(W1, W1b, 256 * 512);
  f32_to_bf16<<<(16 * 256 + 255) / 256, 256, 0, stream>>>(W2, W2b, 16 * 256);

  // MLP with WMMA
  mlp_fused<<<(N + 127) / 128, 256, 0, stream>>>(xb, W1b, b1, W2b, b2, h, N);

  // Graph normalization prep
  deg_init   <<<(N + 255) / 256, 256, 0, stream>>>(dinv, N);
  deg_accum  <<<(E + 255) / 256, 256, 0, stream>>>(ei + E, dinv, E);
  dinv_finish<<<(N + 255) / 256, 256, 0, stream>>>(dinv, N);
  edge_prep  <<<(E + 255) / 256, 256, 0, stream>>>(ei, src32, dst32, norm, dinv, E);

  // K propagation steps, ping-pong zA/zB (initial z = h)
  const long long tN = (long long)N * 16;
  const long long tE = (long long)E * 16;
  const unsigned gN = (unsigned)((tN + 255) / 256);
  const unsigned gE = (unsigned)((tE + 255) / 256);

  const float* cur = h;
  float* bufs[2] = {zA, zB};
  for (int s = 0; s < K_STEPS; ++s) {
    float* nxt = bufs[s & 1];
    prop_init   <<<gN, 256, 0, stream>>>(h, cur, dinv, nxt, N);
    prop_scatter<<<gE, 256, 0, stream>>>(src32, dst32, norm, cur, nxt, E);
    cur = nxt;
  }

  logsoftmax16<<<(N + 255) / 256, 256, 0, stream>>>(cur, out, N);
}
